// EmotionTriadFusion_31078383353877
// MI455X (gfx1250) — matrix operand order repaired and checked by reading, not compiled, not run on hardware
//
#include <hip/hip_runtime.h>
#include <cmath>

// ---------------------------------------------------------------------------
// EmotionTriadFusion for MI455X (gfx1250, wave32, WMMA).
// Big GEMMs use v_wmma_f32_16x16x32_bf16 with double-buffered LDS fed by
// CDNA5 async global->LDS DMA (global_load_async_to_lds_b128 / ASYNCcnt).
// Small per-row ops are VALU kernels.
// ---------------------------------------------------------------------------

typedef __attribute__((ext_vector_type(16))) __bf16 v16bf;
typedef __attribute__((ext_vector_type(8)))  float  v8f;
typedef __attribute__((ext_vector_type(4)))  int    v4i;   // POD 16-byte chunk

#define B_ROWS 16384
#define F_DIM  1024
#define E_DIM  28
#define T_DIM  128

__device__ __forceinline__ float gelu_erf(float x) {
  return 0.5f * x * (1.0f + erff(x * 0.70710678118654752f));
}

// Generic pointers into the LDS aperture carry the LDS byte offset in their
// low 32 bits -> truncation yields the DS address for the async VDST operand.
__device__ __forceinline__ void async_b128(unsigned lds, const void* g) {
  asm volatile("global_load_async_to_lds_b128 %0, %1, off"
               :: "v"(lds), "v"((unsigned long long)(uintptr_t)g) : "memory");
}
__device__ __forceinline__ void wait_async0() {
  asm volatile("s_wait_asynccnt 0x0" ::: "memory");
}

// ---- dst(N,K) bf16  <-  src(K,N) f32 (transpose + cast), one elem/thread ----
__global__ void k_transpose_cast(const float* __restrict__ src,
                                 __bf16* __restrict__ dst, int K, int N) {
  long idx = (long)blockIdx.x * blockDim.x + threadIdx.x;
  long total = (long)K * N;
  if (idx >= total) return;
  int n = (int)(idx / K);
  int k = (int)(idx % K);
  dst[idx] = (__bf16)src[(long)k * N + n];
}

// ---- fusion_feat (B,1024) f32 -> left half of gate_in (B,2048) bf16 --------
__global__ void k_cast_gatein(const float* __restrict__ src,
                              __bf16* __restrict__ dst) {
  long idx = (long)blockIdx.x * blockDim.x + threadIdx.x;
  if (idx >= (long)B_ROWS * F_DIM) return;
  long b = idx / F_DIM;
  int  f = (int)(idx % F_DIM);
  dst[b * 2048 + f] = (__bf16)src[idx];
}

// ---- softmax/LN + tiny matvecs -> GELU token embeddings (B, 3*128) bf16 ----
__global__ __launch_bounds__(128)
void k_embed(const float* __restrict__ el, const float* __restrict__ il,
             const float* __restrict__ av,
             const float* __restrict__ g_e, const float* __restrict__ b_e,
             const float* __restrict__ We,  const float* __restrict__ be,
             const float* __restrict__ g_i, const float* __restrict__ b_i,
             const float* __restrict__ Wi,  const float* __restrict__ bi,
             const float* __restrict__ g_a, const float* __restrict__ b_a,
             const float* __restrict__ Wa,  const float* __restrict__ ba,
             __bf16* __restrict__ tok) {
  __shared__ float se[E_DIM], si[E_DIM], sa[2];
  int b = blockIdx.x, t = threadIdx.x;
  if (t < E_DIM) { se[t] = el[(long)b * E_DIM + t]; si[t] = il[(long)b * E_DIM + t]; }
  if (t < 2)     sa[t] = av[(long)b * 2 + t];
  __syncthreads();
  if (t == 0 || t == 1) {              // softmax + LayerNorm (28 elems, serial)
    float* s        = (t == 0) ? se  : si;
    const float* g  = (t == 0) ? g_e : g_i;
    const float* bb = (t == 0) ? b_e : b_i;
    float mx = -1e30f;
    for (int k = 0; k < E_DIM; k++) mx = fmaxf(mx, s[k]);
    float sum = 0.f;
    for (int k = 0; k < E_DIM; k++) { float e = expf(s[k] - mx); s[k] = e; sum += e; }
    float inv = 1.f / sum, m = 0.f;
    for (int k = 0; k < E_DIM; k++) { s[k] *= inv; m += s[k]; }
    m *= (1.f / E_DIM);
    float v = 0.f;
    for (int k = 0; k < E_DIM; k++) { float d = s[k] - m; v += d * d; }
    v *= (1.f / E_DIM);
    float r = rsqrtf(v + 1e-5f);
    for (int k = 0; k < E_DIM; k++) s[k] = (s[k] - m) * r * g[k] + bb[k];
  } else if (t == 2) {                 // LayerNorm over 2 affect dims
    float m = 0.5f * (sa[0] + sa[1]);
    float d0 = sa[0] - m, d1 = sa[1] - m;
    float r = rsqrtf(0.5f * (d0 * d0 + d1 * d1) + 1e-5f);
    sa[0] = d0 * r * g_a[0] + b_a[0];
    sa[1] = d1 * r * g_a[1] + b_a[1];
  }
  __syncthreads();
  float ae = be[t], ai = bi[t];
  for (int k = 0; k < E_DIM; k++) {
    ae += se[k] * We[k * T_DIM + t];
    ai += si[k] * Wi[k * T_DIM + t];
  }
  float aa = ba[t] + sa[0] * Wa[t] + sa[1] * Wa[T_DIM + t];
  long base = (long)b * 384;
  tok[base + t]       = (__bf16)gelu_erf(ae);
  tok[base + 128 + t] = (__bf16)gelu_erf(ai);
  tok[base + 256 + t] = (__bf16)gelu_erf(aa);
}

// ---------------------------------------------------------------------------
// WMMA GEMM: C(M,N) = act(A(M,K) @ W(K,N) + bias).
// A: bf16 row-major (lda).  Bt: weight pre-transposed to (N,K) bf16 row-major.
// Block tile 128x128, BK=32, 256 threads = 8 waves; wave -> 32(M) x 64(N),
// i.e. 2x4 fragments of v_wmma_f32_16x16x32_bf16.
// Double-buffered LDS filled by async global->LDS DMA: tile i+1 streams in
// (ASYNCcnt) while tile i is consumed by the WMMAs.
// ACT: 0=none, 1=GELU(erf), 2=sigmoid. OUTF/OUTB: compile-time output modes.
// ---------------------------------------------------------------------------
#define LDSP 40            // padded row stride (bf16): 80B -> bank-spread, 16B aligned
#define BUFE (128 * LDSP)  // elements per buffer
#define BUFB (BUFE * 2)    // bytes per buffer

template <int ACT, bool OUTF, bool OUTB>
__global__ __launch_bounds__(256)
void k_wmma_gemm(const __bf16* __restrict__ A, int lda,
                 const __bf16* __restrict__ Bt, int K,
                 const float* __restrict__ bias,
                 float* __restrict__ Cf, int ldcf,
                 __bf16* __restrict__ Cb, int ldcb) {
  __shared__ __bf16 As[2 * BUFE];
  __shared__ __bf16 Bs[2 * BUFE];
  const int t = threadIdx.x;
  const int lane = t & 31;
  const int wave = t >> 5;
  const int wm = wave & 3;        // 0..3 -> M offset wm*32
  const int wn = wave >> 2;       // 0..1 -> N offset wn*64
  const int l16 = lane & 15;
  const bool hi = lane >= 16;
  const int blockM = blockIdx.x * 128;
  const int blockN = blockIdx.y * 128;

  // global tile sources: 128x32 bf16 = 512 16B-chunks; thread owns chunk t, t+256
  const int rowa = t >> 2, qa = t & 3;
  const __bf16* gA0 = A  + (long)(blockM + rowa)      * lda + qa * 8;
  const __bf16* gA1 = A  + (long)(blockM + rowa + 64) * lda + qa * 8;
  const __bf16* gB0 = Bt + (long)(blockN + rowa)      * K   + qa * 8;
  const __bf16* gB1 = Bt + (long)(blockN + rowa + 64) * K   + qa * 8;
  // LDS destinations (buffer 0); buffer 1 = +BUFB bytes
  const unsigned lA0 = (unsigned)(uintptr_t)&As[rowa * LDSP + qa * 8];
  const unsigned lA1 = lA0 + 64 * LDSP * 2;
  const unsigned lB0 = (unsigned)(uintptr_t)&Bs[rowa * LDSP + qa * 8];
  const unsigned lB1 = lB0 + 64 * LDSP * 2;

  // fragment read pointers (ISA 7.12.2 bf16 layouts), buffer 0; +cur*BUFE later
  // A 16x32: lane<16 -> K chunks {0..7,16..23}; lane>=16 -> {8..15,24..31}
  // B 32x16 (from (N,K) LDS tile): lane<16 -> K {0..7,16..23} of col l16; lane>=16 -> {8..15,24..31}
  const __bf16* fragA[2];
  const __bf16* fragB[4];
#pragma unroll
  for (int mi = 0; mi < 2; mi++)
    fragA[mi] = &As[(wm * 32 + mi * 16 + l16) * LDSP + (hi ? 8 : 0)];
#pragma unroll
  for (int ni = 0; ni < 4; ni++)
    fragB[ni] = &Bs[(wn * 64 + ni * 16 + l16) * LDSP + (hi ? 8 : 0)];

  v8f acc[2][4];
#pragma unroll
  for (int mi = 0; mi < 2; mi++)
#pragma unroll
    for (int ni = 0; ni < 4; ni++)
#pragma unroll
      for (int r = 0; r < 8; r++) acc[mi][ni][r] = 0.f;

  // prologue: stream first K-tile into buffer 0
  async_b128(lA0, gA0);
  async_b128(lA1, gA1);
  async_b128(lB0, gB0);
  async_b128(lB1, gB1);

  for (int kt = 0; kt < K; kt += 32) {
    const int cur = (kt >> 5) & 1;
    wait_async0();           // this wave's tile-i DMA landed in LDS
    __syncthreads();         // all waves' DMA landed; prev reads of other buf done

    const int ktn = kt + 32;
    if (ktn < K) {           // stream tile i+1 into the alternate buffer
      const unsigned o = (cur ^ 1) * (unsigned)BUFB;
      async_b128(lA0 + o, gA0 + ktn);
      async_b128(lA1 + o, gA1 + ktn);
      async_b128(lB0 + o, gB0 + ktn);
      async_b128(lB1 + o, gB1 + ktn);
      // speculative prefetch one more tile ahead (global_prefetch_b8)
      __builtin_prefetch(gA0 + ktn + 32, 0, 1);
      __builtin_prefetch(gB0 + ktn + 32, 0, 1);
    }

    const int bo = cur * BUFE;   // element offset of current buffer
    v16bf af[2], bfr[4];
#pragma unroll
    for (int mi = 0; mi < 2; mi++) {
      union { v4i q[2]; v16bf v; } u;
      u.q[0] = *(const v4i*)(fragA[mi] + bo);
      u.q[1] = *(const v4i*)(fragA[mi] + bo + 16);
      af[mi] = u.v;
    }
#pragma unroll
    for (int ni = 0; ni < 4; ni++) {
      union { v4i q[2]; v16bf v; } u;
      u.q[0] = *(const v4i*)(fragB[ni] + bo);
      u.q[1] = *(const v4i*)(fragB[ni] + bo + 16);
      bfr[ni] = u.v;
    }
#pragma unroll
    for (int mi = 0; mi < 2; mi++)
#pragma unroll
      for (int ni = 0; ni < 4; ni++)
        acc[mi][ni] = __builtin_amdgcn_wmma_f32_16x16x32_bf16(
            false, af[mi], false, bfr[ni], (short)0, acc[mi][ni], false, false);
  }

  // epilogue: C 16x16 f32 layout -> row = r + (lane<16?0:8), col = lane&15
  const int colBase = blockN + wn * 64;
#pragma unroll
  for (int mi = 0; mi < 2; mi++) {
    const int rowBase = blockM + wm * 32 + mi * 16 + (hi ? 8 : 0);
#pragma unroll
    for (int ni = 0; ni < 4; ni++) {
      const v8f a = acc[mi][ni];                 // keep in VGPRs, const idx below
      const int col = colBase + ni * 16 + l16;
      const float bv = bias[col];
      float*  pf = OUTF ? (Cf + (long)rowBase * ldcf + col) : nullptr;
      __bf16* pb = OUTB ? (Cb + (long)rowBase * ldcb + col) : nullptr;
#pragma unroll
      for (int r = 0; r < 8; r++) {
        float v = a[r] + bv;
        if (ACT == 1) v = gelu_erf(v);
        else if (ACT == 2) v = 1.f / (1.f + expf(-v));
        if (OUTF) pf[(long)r * ldcf] = v;
        if (OUTB) pb[(long)r * ldcb] = (__bf16)v;
      }
    }
  }
}

// ---- tiny 1-query x 3-key MHA: one block/row, one wave/head (HD=32=wave32) --
__global__ __launch_bounds__(128)
void k_attention(const float* __restrict__ qh, const __bf16* __restrict__ kh,
                 const __bf16* __restrict__ vh,
                 float* __restrict__ attnw, __bf16* __restrict__ attnc) {
  int b = blockIdx.x, t = threadIdx.x;     // t = h*32 + d
  int h = t >> 5;
  float q = qh[(long)b * 128 + t];
  long base = (long)b * 3 * 128 + t;
  float k0 = (float)kh[base], k1 = (float)kh[base + 128], k2 = (float)kh[base + 256];
  float v0 = (float)vh[base], v1 = (float)vh[base + 128], v2 = (float)vh[base + 256];
  float p0 = q * k0, p1 = q * k1, p2 = q * k2;
#pragma unroll
  for (int m = 16; m >= 1; m >>= 1) {      // wave32 butterfly reduce per head
    p0 += __shfl_xor(p0, m);
    p1 += __shfl_xor(p1, m);
    p2 += __shfl_xor(p2, m);
  }
  const float sc = 0.17677669529663687f;   // 1/sqrt(32)
  p0 *= sc; p1 *= sc; p2 *= sc;
  float mx = fmaxf(p0, fmaxf(p1, p2));
  float e0 = expf(p0 - mx), e1 = expf(p1 - mx), e2 = expf(p2 - mx);
  float inv = 1.f / (e0 + e1 + e2);
  float w0 = e0 * inv, w1 = e1 * inv, w2 = e2 * inv;
  __shared__ float lw[4][3];
  if ((t & 31) == 0) { lw[h][0] = w0; lw[h][1] = w1; lw[h][2] = w2; }
  __syncthreads();
  if (t < 3) attnw[(long)b * 3 + t] = 0.25f * (lw[0][t] + lw[1][t] + lw[2][t] + lw[3][t]);
  attnc[(long)b * 128 + t] = (__bf16)(w0 * v0 + w1 * v1 + w2 * v2);
}

// ---- refined = LN(fusion + gate*context), one block/row -------------------
__global__ __launch_bounds__(256)
void k_refined(const float* __restrict__ fusion, const float* __restrict__ gate,
               const float* __restrict__ ctx, const float* __restrict__ g,
               const float* __restrict__ bb, float* __restrict__ out) {
  int b = blockIdx.x, t = threadIdx.x;
  long base = (long)b * F_DIM;
  float x[4], s = 0.f;
#pragma unroll
  for (int i = 0; i < 4; i++) {
    int f = t + i * 256;
    float v = fusion[base + f] + gate[base + f] * ctx[base + f];
    x[i] = v; s += v;
  }
  __shared__ float red[256];
  red[t] = s; __syncthreads();
  for (int o = 128; o > 0; o >>= 1) { if (t < o) red[t] += red[t + o]; __syncthreads(); }
  float mean = red[0] * (1.f / F_DIM);
  __syncthreads();
  float sq = 0.f;
#pragma unroll
  for (int i = 0; i < 4; i++) { float d = x[i] - mean; sq += d * d; }
  red[t] = sq; __syncthreads();
  for (int o = 128; o > 0; o >>= 1) { if (t < o) red[t] += red[t + o]; __syncthreads(); }
  float rinv = rsqrtf(red[0] * (1.f / F_DIM) + 1e-5f);
#pragma unroll
  for (int i = 0; i < 4; i++) {
    int f = t + i * 256;
    out[base + f] = (x[i] - mean) * rinv * g[f] + bb[f];
  }
}

// ---------------------------------------------------------------------------
extern "C" void kernel_launch(void* const* d_in, const int* in_sizes, int n_in,
                              void* d_out, int out_size, void* d_ws, size_t ws_size,
                              hipStream_t stream) {
  const float* fusion   = (const float*)d_in[0];
  const float* el       = (const float*)d_in[1];
  const float* il       = (const float*)d_in[2];
  const float* av       = (const float*)d_in[3];
  const float* ln_exp_g = (const float*)d_in[4];
  const float* ln_exp_b = (const float*)d_in[5];
  const float* W_exp    = (const float*)d_in[6];
  const float* b_exp    = (const float*)d_in[7];
  const float* ln_imp_g = (const float*)d_in[8];
  const float* ln_imp_b = (const float*)d_in[9];
  const float* W_imp    = (const float*)d_in[10];
  const float* b_imp    = (const float*)d_in[11];
  const float* ln_aff_g = (const float*)d_in[12];
  const float* ln_aff_b = (const float*)d_in[13];
  const float* W_aff    = (const float*)d_in[14];
  const float* b_aff    = (const float*)d_in[15];
  const float* Wq_in    = (const float*)d_in[16];
  const float* bq_in    = (const float*)d_in[17];
  const float* Wq       = (const float*)d_in[18];
  const float* bq       = (const float*)d_in[19];
  const float* Wk       = (const float*)d_in[20];
  const float* bk       = (const float*)d_in[21];
  const float* Wv       = (const float*)d_in[22];
  const float* bv       = (const float*)d_in[23];
  const float* Wo       = (const float*)d_in[24];
  const float* bo       = (const float*)d_in[25];
  const float* Wc       = (const float*)d_in[26];
  const float* bc       = (const float*)d_in[27];
  const float* Ws       = (const float*)d_in[28];
  const float* bs       = (const float*)d_in[29];
  const float* Wg1      = (const float*)d_in[30];
  const float* bg1      = (const float*)d_in[31];
  const float* Wg2      = (const float*)d_in[32];
  const float* bg2      = (const float*)d_in[33];
  const float* ln_out_g = (const float*)d_in[34];
  const float* ln_out_b = (const float*)d_in[35];

  float* out_refined = (float*)d_out;
  float* out_attnw   = out_refined + (long)B_ROWS * F_DIM;
  float* out_summary = out_attnw + (long)B_ROWS * 3;

  char* w = (char*)d_ws;
  size_t off = 0;
  auto alloc = [&](size_t bytes) -> void* {
    off = (off + 255) & ~(size_t)255;
    void* p = w + off;
    off += bytes;
    return p;
  };
  // transposed bf16 weights (N,K)
  __bf16* WqinT = (__bf16*)alloc((size_t)128  * 1024 * 2);
  __bf16* WqT   = (__bf16*)alloc((size_t)128  * 128  * 2);
  __bf16* WkT   = (__bf16*)alloc((size_t)128  * 128  * 2);
  __bf16* WvT   = (__bf16*)alloc((size_t)128  * 128  * 2);
  __bf16* WoT   = (__bf16*)alloc((size_t)128  * 128  * 2);
  __bf16* WcT   = (__bf16*)alloc((size_t)1024 * 128  * 2);
  __bf16* WsT   = (__bf16*)alloc((size_t)1024 * 384  * 2);
  __bf16* Wg1T  = (__bf16*)alloc((size_t)1024 * 2048 * 2);
  __bf16* Wg2T  = (__bf16*)alloc((size_t)1024 * 1024 * 2);
  // activations
  __bf16* tok     = (__bf16*)alloc((size_t)B_ROWS * 384 * 2);   // (B,3,128) == (3B,128)
  __bf16* gate_in = (__bf16*)alloc((size_t)B_ROWS * 2048 * 2);  // [fusion | context]
  __bf16* q_bf    = (__bf16*)alloc((size_t)B_ROWS * 128 * 2);
  float*  qh_f    = (float*) alloc((size_t)B_ROWS * 128 * 4);
  __bf16* kh_bf   = (__bf16*)alloc((size_t)B_ROWS * 3 * 128 * 2);
  __bf16* vh_bf   = (__bf16*)alloc((size_t)B_ROWS * 3 * 128 * 2);
  __bf16* attnc   = (__bf16*)alloc((size_t)B_ROWS * 128 * 2);
  __bf16* attno   = (__bf16*)alloc((size_t)B_ROWS * 128 * 2);
  float*  ctx_f   = (float*) alloc((size_t)B_ROWS * 1024 * 4);
  __bf16* gateh   = (__bf16*)alloc((size_t)B_ROWS * 1024 * 2);
  float*  gate_f  = (float*) alloc((size_t)B_ROWS * 1024 * 4);

  auto tgrid = [](long n) { return dim3((unsigned)((n + 255) / 256)); };

  // weight prep
  k_transpose_cast<<<tgrid(1024L*128),  256, 0, stream>>>(Wq_in, WqinT, 1024, 128);
  k_transpose_cast<<<tgrid(128L*128),   256, 0, stream>>>(Wq,  WqT,  128, 128);
  k_transpose_cast<<<tgrid(128L*128),   256, 0, stream>>>(Wk,  WkT,  128, 128);
  k_transpose_cast<<<tgrid(128L*128),   256, 0, stream>>>(Wv,  WvT,  128, 128);
  k_transpose_cast<<<tgrid(128L*128),   256, 0, stream>>>(Wo,  WoT,  128, 128);
  k_transpose_cast<<<tgrid(128L*1024),  256, 0, stream>>>(Wc,  WcT,  128, 1024);
  k_transpose_cast<<<tgrid(384L*1024),  256, 0, stream>>>(Ws,  WsT,  384, 1024);
  k_transpose_cast<<<tgrid(2048L*1024), 256, 0, stream>>>(Wg1, Wg1T, 2048, 1024);
  k_transpose_cast<<<tgrid(1024L*1024), 256, 0, stream>>>(Wg2, Wg2T, 1024, 1024);
  k_cast_gatein<<<tgrid((long)B_ROWS * F_DIM), 256, 0, stream>>>(fusion, gate_in);

  // token embeddings
  k_embed<<<dim3(B_ROWS), 128, 0, stream>>>(
      el, il, av, ln_exp_g, ln_exp_b, W_exp, b_exp,
      ln_imp_g, ln_imp_b, W_imp, b_imp, ln_aff_g, ln_aff_b, W_aff, b_aff, tok);

  const dim3 gN128(B_ROWS / 128, 1);       // M=16384, N=128
  const dim3 gN128x3(3 * B_ROWS / 128, 1); // M=49152, N=128
  const dim3 gN1024(B_ROWS / 128, 8);      // M=16384, N=1024

  // q = fusion @ Wq_in + bq_in  (bf16 out)
  k_wmma_gemm<0, false, true><<<gN128, 256, 0, stream>>>(
      gate_in, 2048, WqinT, 1024, bq_in, (float*)nullptr, 0, q_bf, 128);
  // qh = q @ Wq + bq  (f32 out)
  k_wmma_gemm<0, true, false><<<gN128, 256, 0, stream>>>(
      q_bf, 128, WqT, 128, bq, qh_f, 128, (__bf16*)nullptr, 0);
  // kh/vh = tokens @ Wk/Wv + b  (on (3B,128) view)
  k_wmma_gemm<0, false, true><<<gN128x3, 256, 0, stream>>>(
      tok, 128, WkT, 128, bk, (float*)nullptr, 0, kh_bf, 128);
  k_wmma_gemm<0, false, true><<<gN128x3, 256, 0, stream>>>(
      tok, 128, WvT, 128, bv, (float*)nullptr, 0, vh_bf, 128);
  // attention + head-mean weights
  k_attention<<<dim3(B_ROWS), 128, 0, stream>>>(qh_f, kh_bf, vh_bf, out_attnw, attnc);
  // attn_out = attnc @ Wo + bo
  k_wmma_gemm<0, false, true><<<gN128, 256, 0, stream>>>(
      attnc, 128, WoT, 128, bo, (float*)nullptr, 0, attno, 128);
  // context = attn_out @ Wc + bc   (f32 + bf16 into gate_in right half)
  k_wmma_gemm<0, true, true><<<gN1024, 256, 0, stream>>>(
      attno, 128, WcT, 128, bc, ctx_f, 1024, gate_in + 1024, 2048);
  // gate_h = gelu([fusion|context] @ Wg1 + bg1)
  k_wmma_gemm<1, false, true><<<gN1024, 256, 0, stream>>>(
      gate_in, 2048, Wg1T, 2048, bg1, (float*)nullptr, 0, gateh, 1024);
  // gate = sigmoid(gate_h @ Wg2 + bg2)
  k_wmma_gemm<2, true, false><<<gN1024, 256, 0, stream>>>(
      gateh, 1024, Wg2T, 1024, bg2, gate_f, 1024, (__bf16*)nullptr, 0);
  // refined = LN(fusion + gate*context)
  k_refined<<<dim3(B_ROWS), 256, 0, stream>>>(fusion, gate_f, ctx_f,
                                              ln_out_g, ln_out_b, out_refined);
  // summary = gelu([exp|imp|aff] @ Ws + bs)
  k_wmma_gemm<1, true, false><<<gN1024, 256, 0, stream>>>(
      tok, 384, WsT, 384, bs, out_summary, 1024, (__bf16*)nullptr, 0);
}
